// EncoderDecoder_69320772157638
// MI455X (gfx1250) — matrix operand order repaired and checked by reading, not compiled, not run on hardware
//
#include <hip/hip_runtime.h>
#include <hip/hip_bf16.h>

#define V_ 32000
#define E_ 512
#define H_ 1024
#define T_ 4096
#define G_ 4096   // 4*H

typedef __attribute__((ext_vector_type(16))) __bf16 v16bf;
typedef __attribute__((ext_vector_type(8)))  float  v8f;
typedef __attribute__((ext_vector_type(8)))  unsigned short ushort8;

union Frag { v16bf v; ushort8 h[2]; };

__device__ __forceinline__ unsigned short f2bf(float f) {
  unsigned int u = __float_as_uint(f);
  u += 0x7FFFu + ((u >> 16) & 1u);      // round-to-nearest-even
  return (unsigned short)(u >> 16);
}

__device__ __forceinline__ float sigmoidf_(float x) {
  return 1.0f / (1.0f + __expf(-x));
}

// ---------------------------------------------------------------- init
__global__ void k_init(int* bar, float* h, float* c, unsigned short* hs_pad_row) {
  int j = threadIdx.x;
  if (j < 4) bar[j] = 0;
  h[j] = 0.f;
  c[j] = 0.f;
  hs_pad_row[j] = 0;   // zero the padded 4096th row of hs (bf16)
}

// ------------------------------------------------------- f32 -> bf16
__global__ void k_f32_to_bf16(const float* __restrict__ in,
                              unsigned short* __restrict__ out, size_t n) {
  size_t i = (size_t)blockIdx.x * blockDim.x + threadIdx.x;
  if (i < n) out[i] = f2bf(in[i]);
}

// --------------------------------------------- embedding gather -> bf16
__global__ void k_gather_embed(const int* __restrict__ seq,
                               const float* __restrict__ emb,
                               unsigned short* __restrict__ out,
                               int valid_rows, int rows_stored) {
  size_t i = (size_t)blockIdx.x * blockDim.x + threadIdx.x;
  size_t total = (size_t)rows_stored * E_;
  if (i >= total) return;
  int row = (int)(i / E_);
  int e   = (int)(i % E_);
  float v = 0.f;
  if (row < valid_rows) v = emb[(size_t)seq[row] * E_ + e];
  out[i] = f2bf(v);
}

// ---------------------------------------------------------- WMMA GEMM
// C[M,N] = A[M,K] * W[N,K]^T (+bias0[n]) (+bias1[n]); bf16 in, f32 out.
// Each wave computes a 16x64 strip (1 M-tile x 4 N-tiles): the A fragment
// is loaded once per K-step and reused across 4 WMMAs (40B loaded / WMMA
// instead of 64B). Grids are exact multiples of 8 waves, so EXEC is
// all-ones around every v_wmma.
__global__ void k_wmma_gemm(const unsigned short* __restrict__ A,  // [Mt*16, K]
                            const unsigned short* __restrict__ W,  // [Ntg*64, K]
                            const float* __restrict__ bias0,
                            const float* __restrict__ bias1,
                            float* __restrict__ C,                 // [Mvalid, N]
                            int Mt, int Ntg, int K, int Mvalid, int N,
                            int nt_store) {
  const int lane = threadIdx.x & 31;
  const int wave = threadIdx.x >> 5;
  const int grp  = blockIdx.x * (blockDim.x >> 5) + wave;
  if (grp >= Mt * Ntg) return;               // wave-uniform guard
  const int mt = grp / Ntg;
  const int ng = grp % Ntg;
  const int m0 = mt << 4, n0 = ng << 6;      // 16 rows x 64 cols
  const int l16 = lane & 15;
  const int hi  = lane >> 4;                 // half-wave select

  // A fragment: row m0+l16; 16 consecutive K at k0+hi*16
  const unsigned short* arow = A + (size_t)(m0 + l16) * K + hi * 16;
  // B fragments from W[N,K]: row n0+j*16+l16; 16 consecutive K at k0+hi*16
  const unsigned short* wrow0 = W + (size_t)(n0 +  0 + l16) * K + hi * 16;
  const unsigned short* wrow1 = W + (size_t)(n0 + 16 + l16) * K + hi * 16;
  const unsigned short* wrow2 = W + (size_t)(n0 + 32 + l16) * K + hi * 16;
  const unsigned short* wrow3 = W + (size_t)(n0 + 48 + l16) * K + hi * 16;

  v8f acc0 = {}, acc1 = {}, acc2 = {}, acc3 = {};
  for (int k0 = 0; k0 < K; k0 += 32) {
    Frag a, b0, b1, b2, b3;
    a.h[0]  = *(const ushort8*)(arow  + k0);
    a.h[1]  = *(const ushort8*)(arow  + k0 + 8);
    b0.h[0] = *(const ushort8*)(wrow0 + k0);
    b0.h[1] = *(const ushort8*)(wrow0 + k0 + 8);
    b1.h[0] = *(const ushort8*)(wrow1 + k0);
    b1.h[1] = *(const ushort8*)(wrow1 + k0 + 8);
    b2.h[0] = *(const ushort8*)(wrow2 + k0);
    b2.h[1] = *(const ushort8*)(wrow2 + k0 + 8);
    b3.h[0] = *(const ushort8*)(wrow3 + k0);
    b3.h[1] = *(const ushort8*)(wrow3 + k0 + 8);
    __builtin_prefetch(arow  + k0 + 128, 0, 3);
    __builtin_prefetch(wrow0 + k0 + 128, 0, 3);
    __builtin_prefetch(wrow1 + k0 + 128, 0, 3);
    __builtin_prefetch(wrow2 + k0 + 128, 0, 3);
    __builtin_prefetch(wrow3 + k0 + 128, 0, 3);
    acc0 = __builtin_amdgcn_wmma_f32_16x16x32_bf16(
        false, a.v, false, b0.v, (short)0, acc0, false, false);
    acc1 = __builtin_amdgcn_wmma_f32_16x16x32_bf16(
        false, a.v, false, b1.v, (short)0, acc1, false, false);
    acc2 = __builtin_amdgcn_wmma_f32_16x16x32_bf16(
        false, a.v, false, b2.v, (short)0, acc2, false, false);
    acc3 = __builtin_amdgcn_wmma_f32_16x16x32_bf16(
        false, a.v, false, b3.v, (short)0, acc3, false, false);
  }

  // C/D layout: VGPR r -> row m0+r (+8 for hi lanes), col = tile_n0 + l16
#pragma unroll
  for (int j = 0; j < 4; ++j) {
    const v8f acc = (j == 0) ? acc0 : (j == 1) ? acc1 : (j == 2) ? acc2 : acc3;
    const int col = n0 + j * 16 + l16;
    float bias = 0.f;
    if (bias0) bias += bias0[col];
    if (bias1) bias += bias1[col];
    if (nt_store) {
#pragma unroll
      for (int r = 0; r < 8; ++r) {
        int row = m0 + r + hi * 8;
        if (row < Mvalid)
          __builtin_nontemporal_store(acc[r] + bias, &C[(size_t)row * N + col]);
      }
    } else {
#pragma unroll
      for (int r = 0; r < 8; ++r) {
        int row = m0 + r + hi * 8;
        if (row < Mvalid) C[(size_t)row * N + col] = acc[r] + bias;
      }
    }
  }
}

// ------------------------------------------------ persistent LSTM scan
__device__ __forceinline__ void gsync(volatile int* bar, int nblk) {
  __threadfence();
  __syncthreads();
  if (threadIdx.x == 0) {
    int gen = bar[1];
    if (atomicAdd((int*)bar, 1) == nblk - 1) {
      bar[0] = 0;
      __threadfence();
      atomicAdd((int*)bar + 1, 1);
    } else {
      while (bar[1] == gen) __builtin_amdgcn_s_sleep(2);
    }
  }
  __syncthreads();
}

__global__ void k_lstm_seq(const float* __restrict__ Gx,   // [Tsteps, 4H] = Wih@x + bih + bhh
                           const float* __restrict__ Whh,  // [4H, H] f32 (L2-resident)
                           float* __restrict__ h,
                           float* __restrict__ c,
                           float* __restrict__ gates,      // [4H] scratch
                           unsigned short* __restrict__ hs_bf16, // [T,H] bf16 or nullptr
                           int Tsteps, int* bar) {
  const int lane = threadIdx.x & 31;
  const int wave = threadIdx.x >> 5;
  const int nw   = (gridDim.x * blockDim.x) >> 5;
  const int gw   = ((blockIdx.x * blockDim.x) >> 5) + wave;
  const int gtid = blockIdx.x * blockDim.x + threadIdx.x;

  for (int t = 0; t < Tsteps; ++t) {
    const float* gx = Gx + (size_t)t * G_;
    // phase 1: gates[r] = Gx[t,r] + Whh[r,:] . h   (one wave per row)
    for (int r = gw; r < G_; r += nw) {
      const float4* wr = (const float4*)(Whh + (size_t)r * H_);
      const float4* hv = (const float4*)h;
      float s = 0.f;
#pragma unroll 4
      for (int kk = lane; kk < H_ / 4; kk += 32) {
        float4 w = wr[kk], x = hv[kk];
        s = fmaf(w.x, x.x, s); s = fmaf(w.y, x.y, s);
        s = fmaf(w.z, x.z, s); s = fmaf(w.w, x.w, s);
      }
      for (int off = 16; off; off >>= 1) s += __shfl_down(s, off, 32);
      if (lane == 0) gates[r] = gx[r] + s;
    }
    gsync(bar, gridDim.x);
    // phase 2: cell update (i,f,g,o gate order)
    if (gtid < H_) {
      int j = gtid;
      float ig = sigmoidf_(gates[j]);
      float fg = sigmoidf_(gates[H_ + j]);
      float gg = tanhf(gates[2 * H_ + j]);
      float og = sigmoidf_(gates[3 * H_ + j]);
      float cn = fmaf(fg, c[j], ig * gg);
      float hn = og * tanhf(cn);
      c[j] = cn;
      h[j] = hn;
      if (hs_bf16) hs_bf16[(size_t)t * H_ + j] = f2bf(hn);
    }
    gsync(bar, gridDim.x);
  }
}

// ---------------------------------------------------------------- host
extern "C" void kernel_launch(void* const* d_in, const int* in_sizes, int n_in,
                              void* d_out, int out_size, void* d_ws, size_t ws_size,
                              hipStream_t stream) {
  (void)in_sizes; (void)n_in; (void)out_size; (void)ws_size;
  const int*   seq     = (const int*)  d_in[0];
  const float* enc_emb = (const float*)d_in[1];
  const float* enc_Wih = (const float*)d_in[2];
  const float* enc_Whh = (const float*)d_in[3];
  const float* enc_bih = (const float*)d_in[4];
  const float* enc_bhh = (const float*)d_in[5];
  const float* dec_emb = (const float*)d_in[6];
  const float* dec_Wih = (const float*)d_in[7];
  const float* dec_Whh = (const float*)d_in[8];
  const float* dec_bih = (const float*)d_in[9];
  const float* dec_bhh = (const float*)d_in[10];
  const float* out_W   = (const float*)d_in[11];
  const float* out_b   = (const float*)d_in[12];
  float* logits = (float*)d_out;                 // [T-1, 1, V]

  char* ws = (char*)d_ws;
  size_t off = 0;
  auto alloc = [&](size_t bytes) -> void* {
    void* p = ws + off;
    off += (bytes + 255) & ~(size_t)255;
    return p;
  };
  int*            bar   = (int*)alloc(16);
  float*          h     = (float*)alloc(H_ * 4);
  float*          c     = (float*)alloc(H_ * 4);
  float*          gates = (float*)alloc(G_ * 4);
  unsigned short* Xe    = (unsigned short*)alloc((size_t)T_ * E_ * 2);
  unsigned short* Xd    = (unsigned short*)alloc((size_t)T_ * E_ * 2);
  unsigned short* WihE  = (unsigned short*)alloc((size_t)G_ * E_ * 2);
  unsigned short* WihD  = (unsigned short*)alloc((size_t)G_ * E_ * 2);
  unsigned short* Wout  = (unsigned short*)alloc((size_t)V_ * H_ * 2);
  float*          GxE   = (float*)alloc((size_t)T_ * G_ * 4);
  float*          GxD   = (float*)alloc((size_t)T_ * G_ * 4);
  unsigned short* hsB   = (unsigned short*)alloc((size_t)T_ * H_ * 2);

  k_init<<<1, 1024, 0, stream>>>(bar, h, c, hsB + (size_t)(T_ - 1) * H_);

  auto cvt = [&](const float* in, unsigned short* out, size_t n) {
    int blk = (int)((n + 255) / 256);
    k_f32_to_bf16<<<blk, 256, 0, stream>>>(in, out, n);
  };
  cvt(enc_Wih, WihE, (size_t)G_ * E_);
  cvt(dec_Wih, WihD, (size_t)G_ * E_);
  cvt(out_W,   Wout, (size_t)V_ * H_);

  {
    size_t n = (size_t)T_ * E_;
    int blk = (int)((n + 255) / 256);
    k_gather_embed<<<blk, 256, 0, stream>>>(seq, enc_emb, Xe, T_,     T_);
    k_gather_embed<<<blk, 256, 0, stream>>>(seq, dec_emb, Xd, T_ - 1, T_);
  }

  // Hoisted input-side GEMMs: Gx = X @ Wih^T + bih + bhh   [T, 4H]
  // Gx is written once, read once sequentially -> NT stores keep L2 clean.
  {
    int Mt = T_ / 16, Ntg = G_ / 64;         // 256 x 64 strip-groups
    int blocks = (Mt * Ntg + 7) / 8;         // 2048, exact fit (8 waves/blk)
    k_wmma_gemm<<<blocks, 256, 0, stream>>>(Xe, WihE, enc_bih, enc_bhh,
                                            GxE, Mt, Ntg, E_, T_, G_, 1);
    k_wmma_gemm<<<blocks, 256, 0, stream>>>(Xd, WihD, dec_bih, dec_bhh,
                                            GxD, Mt, Ntg, E_, T_, G_, 1);
  }

  // Sequential recurrences (encoder final (h,c) feeds decoder in place)
  k_lstm_seq<<<64, 256, 0, stream>>>(GxE, enc_Whh, h, c, gates, nullptr, T_,     bar);
  k_lstm_seq<<<64, 256, 0, stream>>>(GxD, dec_Whh, h, c, gates, hsB,     T_ - 1, bar);

  // Output projection: logits = hs @ out_W^T + out_b   [4095, 32000]
  // 524MB written once, never re-read -> NT stores; out_W/hs stay in L2.
  {
    int Mt = T_ / 16, Ntg = V_ / 64;         // 256 x 500 strip-groups
    int blocks = (Mt * Ntg + 7) / 8;         // 16000, exact fit
    k_wmma_gemm<<<blocks, 256, 0, stream>>>(hsB, Wout, out_b, nullptr,
                                            logits, Mt, Ntg, H_, T_ - 1, V_, 1);
  }
}
